// SoftAttention_50886772523151
// MI455X (gfx1250) — matrix operand order repaired and checked by reading, not compiled, not run on hardware
//
#include <hip/hip_runtime.h>

// MI455X / gfx1250 fused soft-attention:
//   q = h@qW+qb ; k = msg@kW+kb ; v = relu(msg@vW+vb)
//   scores = (q.k)/4 * m ; ==0 -> -1e6 ; softmax * m ; ctx = p@v
//   out = layernorm(ctx@dW + db + h)
//
// - Projections use V_WMMA_F32_16X16X32_F16 (f16 in, f32 accumulate): the K=32
//   f16 shape keeps matrix math under the 23.3 TB/s HBM stream time.
// - Weights are pre-packed once per launch into fragment-ready f16 layout in
//   d_ws, so each B fragment is 2 coalesced global_load_b128.
// - The dominant message stream (64 KB/block) is moved with
//   GLOBAL_LOAD_ASYNC_TO_LDS_B128 (ASYNCcnt) directly into LDS, overlapping
//   with the h / mask staging loads; f32->f16 conversion happens during A-fragment
//   construction.
// - Attention softmax is spread over all 256 threads (8 lanes per token-head,
//   __shfl_xor reductions); layernorm uses one wave per token.

typedef __attribute__((ext_vector_type(16))) _Float16 v16h;
typedef __attribute__((ext_vector_type(8)))  _Float16 v8h;
typedef __attribute__((ext_vector_type(8)))  float    v8f;

#define HID   128
#define NA    32
#define NH    8
#define HD    16
#define TPB   4        // tokens per workgroup
#define ROWPH 136      // padded f16 LDS row stride (272B)
#define ROWPF 132      // padded f32 LDS row stride (528B)
#define WFRAG 16384    // f16 elements per pre-packed weight matrix

// ---------------------------------------------------------------------------
// Pre-pass: pack W[128][128] f32 into fragment-ready f16 (B-matrix layout:
// lane = column, 2 K-values per VGPR, half h holds K {8h..8h+7, 16+8h..16+8h+7}).
// ---------------------------------------------------------------------------
__global__ __launch_bounds__(256) void pack_weights(
    const float* __restrict__ qW, const float* __restrict__ kW,
    const float* __restrict__ vW, const float* __restrict__ dW,
    _Float16* __restrict__ ws) {
  const int o = blockIdx.x * 256 + threadIdx.x;  // 0 .. 4*WFRAG-1
  const int m    = o >> 14;
  const int r    = o & (WFRAG - 1);
  const int j    = r & 15;
  const int lane = (r >> 4) & 31;
  const int ks   = (r >> 9) & 3;
  const int nt   = r >> 11;
  const float* W = (m == 0) ? qW : (m == 1) ? kW : (m == 2) ? vW : dW;
  const int kk  = ks * 32 + ((j & 8) ? 16 : 0) + ((lane & 16) ? 8 : 0) + (j & 7);
  const int col = nt * 16 + (lane & 15);
  ws[o] = (_Float16)W[kk * HID + col];
}

// B fragment from pre-packed weights: 2 coalesced global_load_b128 per lane.
__device__ __forceinline__ v16h load_bfragw(const _Float16* __restrict__ wsW,
                                            int nt, int ks, int lane) {
  const _Float16* p = wsW + ((((nt << 2) + ks) << 5) + lane) * 16;
  const v8h lo = *(const v8h*)(p);
  const v8h hi = *(const v8h*)(p + 8);
  v16h b;
#pragma unroll
  for (int i = 0; i < 8; ++i) { b[i] = lo[i]; b[i + 8] = hi[i]; }
  return b;
}

// A fragment from f16 LDS (h / ctx): lane l<16 -> row l, K {k0..k0+7,
// k0+16..k0+23}; lane>=16 -> row l-16, K {k0+8..k0+15, k0+24..k0+31}.
__device__ __forceinline__ v16h load_afrag16(const _Float16* base, int row0,
                                             int k0, int lane) {
  const int r  = row0 + (lane & 15);
  const int ka = k0 + ((lane & 16) ? 8 : 0);
  const _Float16* p = base + r * ROWPH + ka;
  const v8h lo = *(const v8h*)(p);       // ds_load_b128
  const v8h hi = *(const v8h*)(p + 16);  // ds_load_b128
  v16h a;
#pragma unroll
  for (int i = 0; i < 8; ++i) { a[i] = lo[i]; a[i + 8] = hi[i]; }
  return a;
}

// A fragment from f32 LDS (async-staged message), converting on the fly.
__device__ __forceinline__ v16h load_afrag32(const float* base, int row0,
                                             int k0, int lane) {
  const int r  = row0 + (lane & 15);
  const int ka = k0 + ((lane & 16) ? 8 : 0);
  const float* p = base + r * ROWPF + ka;
  const float4 x0 = *(const float4*)(p);       // K ka..ka+3
  const float4 x1 = *(const float4*)(p + 4);   // K ka+4..ka+7
  const float4 y0 = *(const float4*)(p + 16);  // K ka+16..ka+19
  const float4 y1 = *(const float4*)(p + 20);  // K ka+20..ka+23
  v16h a;
  a[0]  = (_Float16)x0.x; a[1]  = (_Float16)x0.y;
  a[2]  = (_Float16)x0.z; a[3]  = (_Float16)x0.w;
  a[4]  = (_Float16)x1.x; a[5]  = (_Float16)x1.y;
  a[6]  = (_Float16)x1.z; a[7]  = (_Float16)x1.w;
  a[8]  = (_Float16)y0.x; a[9]  = (_Float16)y0.y;
  a[10] = (_Float16)y0.z; a[11] = (_Float16)y0.w;
  a[12] = (_Float16)y1.x; a[13] = (_Float16)y1.y;
  a[14] = (_Float16)y1.z; a[15] = (_Float16)y1.w;
  return a;
}

__global__ __launch_bounds__(256) void softattn_gfx1250(
    const float* __restrict__ h,   const float* __restrict__ message,
    const float* __restrict__ mask,
    const float* __restrict__ qb,  const float* __restrict__ kb,
    const float* __restrict__ vb,  const float* __restrict__ db,
    const float* __restrict__ lnw, const float* __restrict__ lnb,
    const _Float16* __restrict__ wsW,
    float* __restrict__ out) {
  extern __shared__ char smem[];
  float*    msgS = (float*)   (smem +      0);  // 128 x ROWPF f32 (async tile)
  _Float16* kS   = (_Float16*)(smem +  67584);  // 128 x ROWPH f16
  _Float16* vS   = (_Float16*)(smem + 102400);  // 128 x ROWPH f16
  _Float16* hS   = (_Float16*)(smem + 137216);  //  16 x ROWPH f16
  _Float16* ctxS = (_Float16*)(smem + 141568);  //  16 x ROWPH f16
  float*    qS   = (float*)   (smem + 145920);  // TPB x 128 f32
  float*    hidS = (float*)   (smem + 147968);  // TPB x 128 f32
  float*    mskS = (float*)   (smem + 150016);  // TPB*NA f32
  // total 150528 bytes (< 320KB/WGP, 2 workgroups per WGP)

  const int tid  = threadIdx.x;
  const int lane = tid & 31;
  const int wave = tid >> 5;
  const long gtok0 = (long)blockIdx.x * TPB;

  const _Float16* wsQ = wsW;
  const _Float16* wsK = wsW + WFRAG;
  const _Float16* wsV = wsW + 2 * WFRAG;
  const _Float16* wsD = wsW + 3 * WFRAG;

  // ---- stage: message -> f32 LDS via async-to-LDS DMA (ASYNCcnt) ---------
  {
    const unsigned msgBase = (unsigned)(uintptr_t)msgS;  // LDS byte offset
    const unsigned long long gaddr =
        (unsigned long long)(uintptr_t)(message + gtok0 * (NA * HID));
#pragma unroll
    for (int i = 0; i < (TPB * NA * HID / 4) / 256; ++i) {  // 16 x b128/lane
      const int e4 = tid + i * 256;
      const unsigned ldsa =
          msgBase + (unsigned)((e4 >> 5) * (ROWPF * 4) + (e4 & 31) * 16);
      const unsigned voff = (unsigned)(e4 * 16);
      asm volatile("global_load_async_to_lds_b128 %0, %1, %2"
                   :: "v"(ldsa), "v"(voff), "s"(gaddr) : "memory");
    }
    // h -> f16 LDS, mask -> f32 LDS (small, regular loads)
    if (tid < TPB * HID / 4) {
      const float4 d = ((const float4*)(h + gtok0 * HID))[tid];
      const int e = tid << 2;
      _Float16* p = &hS[(e >> 7) * ROWPH + (e & (HID - 1))];
      p[0] = (_Float16)d.x; p[1] = (_Float16)d.y;
      p[2] = (_Float16)d.z; p[3] = (_Float16)d.w;
    }
    if (tid < TPB * NA) mskS[tid] = mask[gtok0 * NA + tid];
    asm volatile("s_wait_asynccnt 0" ::: "memory");
  }
  __syncthreads();

  const int nt   = wave;                  // N-tile == head index
  const int colL = (nt << 4) + (lane & 15);
  const int rb   = (lane & 16) ? 8 : 0;   // C-tile row base for this half-wave

  // ---- Q = h @ qW + qb (rows >= TPB discarded) ---------------------------
  {
    const float bias = qb[colL];
    v8f acc = {};
#pragma unroll
    for (int ks = 0; ks < 4; ++ks) {
      v16h b = load_bfragw(wsQ, nt, ks, lane);
      v16h a = load_afrag16(hS, 0, ks * 32, lane);
      acc = __builtin_amdgcn_wmma_f32_16x16x32_f16(false, a, false, b,
                                                   (short)0, acc, false, false);
    }
#pragma unroll
    for (int v = 0; v < 8; ++v) {
      const int r = rb + v;
      if (r < TPB) qS[r * HID + colL] = acc[v] + bias;
    }
  }

  // ---- K = message @ kW + kb ---------------------------------------------
  {
    const float bias = kb[colL];
    const v16h b0 = load_bfragw(wsK, nt, 0, lane);
    const v16h b1 = load_bfragw(wsK, nt, 1, lane);
    const v16h b2 = load_bfragw(wsK, nt, 2, lane);
    const v16h b3 = load_bfragw(wsK, nt, 3, lane);
    for (int mt = 0; mt < TPB * NA / 16; ++mt) {
      v8f acc = {};
      acc = __builtin_amdgcn_wmma_f32_16x16x32_f16(false, load_afrag32(msgS, mt * 16,  0, lane), false, b0, (short)0, acc, false, false);
      acc = __builtin_amdgcn_wmma_f32_16x16x32_f16(false, load_afrag32(msgS, mt * 16, 32, lane), false, b1, (short)0, acc, false, false);
      acc = __builtin_amdgcn_wmma_f32_16x16x32_f16(false, load_afrag32(msgS, mt * 16, 64, lane), false, b2, (short)0, acc, false, false);
      acc = __builtin_amdgcn_wmma_f32_16x16x32_f16(false, load_afrag32(msgS, mt * 16, 96, lane), false, b3, (short)0, acc, false, false);
#pragma unroll
      for (int v = 0; v < 8; ++v)
        kS[(mt * 16 + rb + v) * ROWPH + colL] = (_Float16)(acc[v] + bias);
    }
  }

  // ---- V = relu(message @ vW + vb) ---------------------------------------
  {
    const float bias = vb[colL];
    const v16h b0 = load_bfragw(wsV, nt, 0, lane);
    const v16h b1 = load_bfragw(wsV, nt, 1, lane);
    const v16h b2 = load_bfragw(wsV, nt, 2, lane);
    const v16h b3 = load_bfragw(wsV, nt, 3, lane);
    for (int mt = 0; mt < TPB * NA / 16; ++mt) {
      v8f acc = {};
      acc = __builtin_amdgcn_wmma_f32_16x16x32_f16(false, load_afrag32(msgS, mt * 16,  0, lane), false, b0, (short)0, acc, false, false);
      acc = __builtin_amdgcn_wmma_f32_16x16x32_f16(false, load_afrag32(msgS, mt * 16, 32, lane), false, b1, (short)0, acc, false, false);
      acc = __builtin_amdgcn_wmma_f32_16x16x32_f16(false, load_afrag32(msgS, mt * 16, 64, lane), false, b2, (short)0, acc, false, false);
      acc = __builtin_amdgcn_wmma_f32_16x16x32_f16(false, load_afrag32(msgS, mt * 16, 96, lane), false, b3, (short)0, acc, false, false);
#pragma unroll
      for (int v = 0; v < 8; ++v)
        vS[(mt * 16 + rb + v) * ROWPH + colL] = (_Float16)fmaxf(acc[v] + bias, 0.f);
    }
  }
  __syncthreads();

  // ---- softmax attention: 8 lanes per (token,head), 4 neighbors each -----
  {
    const int grp  = tid >> 3;   // 0..31 == tok*NH + head
    const int part = tid & 7;    // 8-way neighbor split (within 8-lane group)
    const int tok  = grp >> 3;
    const int hd   = grp & 7;
    float qv[HD];
    {
      const float4* q4 = (const float4*)&qS[tok * HID + hd * HD];
#pragma unroll
      for (int i = 0; i < 4; ++i) {
        const float4 t = q4[i];
        qv[4 * i] = t.x; qv[4 * i + 1] = t.y;
        qv[4 * i + 2] = t.z; qv[4 * i + 3] = t.w;
      }
    }
    float sc[4];
    float mx = -3.4e38f;
#pragma unroll
    for (int jj = 0; jj < 4; ++jj) {
      const int j = part * 4 + jj;
      const v8h* kr = (const v8h*)&kS[(tok * NA + j) * ROWPH + hd * HD];
      const v8h k0 = kr[0], k1 = kr[1];
      float s = 0.f;
#pragma unroll
      for (int d = 0; d < 8; ++d)
        s += qv[d] * (float)k0[d] + qv[d + 8] * (float)k1[d];
      const float m = mskS[tok * NA + j];
      s = s * 0.25f * m;                 // / sqrt(HEAD_DIM) then * mask
      if (s == 0.f) s = -1000000.0f;     // where(scores==0, NEG_BIG)
      sc[jj] = s;
      mx = fmaxf(mx, s);
    }
#pragma unroll
    for (int m = 1; m < 8; m <<= 1) mx = fmaxf(mx, __shfl_xor(mx, m, 32));
    float sum = 0.f;
#pragma unroll
    for (int jj = 0; jj < 4; ++jj) {
      const float e = __expf(sc[jj] - mx);
      sc[jj] = e; sum += e;
    }
#pragma unroll
    for (int m = 1; m < 8; m <<= 1) sum += __shfl_xor(sum, m, 32);
    const float inv = 1.f / sum;
    float cv[HD];
#pragma unroll
    for (int d = 0; d < HD; ++d) cv[d] = 0.f;
#pragma unroll
    for (int jj = 0; jj < 4; ++jj) {
      const int j = part * 4 + jj;
      const float p = sc[jj] * inv * mskS[tok * NA + j];
      const v8h* vr = (const v8h*)&vS[(tok * NA + j) * ROWPH + hd * HD];
      const v8h a0 = vr[0], a1 = vr[1];
#pragma unroll
      for (int d = 0; d < 8; ++d) {
        cv[d]     += p * (float)a0[d];
        cv[d + 8] += p * (float)a1[d];
      }
    }
#pragma unroll
    for (int m = 1; m < 8; m <<= 1) {
#pragma unroll
      for (int d = 0; d < HD; ++d) cv[d] += __shfl_xor(cv[d], m, 32);
    }
    if (part == 0) {
      _Float16* cp = &ctxS[tok * ROWPH + hd * HD];
#pragma unroll
      for (int d = 0; d < HD; ++d) cp[d] = (_Float16)cv[d];
    }
  }
  __syncthreads();

  // ---- hidden = ctx @ dW + db + h ----------------------------------------
  {
    const float bias = db[colL];
    v8f acc = {};
#pragma unroll
    for (int ks = 0; ks < 4; ++ks) {
      v16h b = load_bfragw(wsD, nt, ks, lane);
      v16h a = load_afrag16(ctxS, 0, ks * 32, lane);
      acc = __builtin_amdgcn_wmma_f32_16x16x32_f16(false, a, false, b,
                                                   (short)0, acc, false, false);
    }
#pragma unroll
    for (int v = 0; v < 8; ++v) {
      const int r = rb + v;
      if (r < TPB)
        hidS[r * HID + colL] = acc[v] + bias + h[(gtok0 + r) * HID + colL];
    }
  }
  __syncthreads();

  // ---- layernorm: one wave per token, 4 columns per lane -----------------
  if (tid < TPB * 32) {
    const int tok = tid >> 5;
    const int ln  = tid & 31;
    const float4 x = *(const float4*)&hidS[tok * HID + ln * 4];
    float s1 = x.x + x.y + x.z + x.w;
    float s2 = x.x * x.x + x.y * x.y + x.z * x.z + x.w * x.w;
#pragma unroll
    for (int m = 1; m < 32; m <<= 1) {
      s1 += __shfl_xor(s1, m, 32);
      s2 += __shfl_xor(s2, m, 32);
    }
    const float u   = s1 * (1.f / HID);
    const float var = s2 * (1.f / HID) - u * u;
    const float inv = rsqrtf(var + 1e-12f);
    const float4 w = *(const float4*)&lnw[ln * 4];
    const float4 b = *(const float4*)&lnb[ln * 4];
    float4 o;
    o.x = w.x * ((x.x - u) * inv) + b.x;
    o.y = w.y * ((x.y - u) * inv) + b.y;
    o.z = w.z * ((x.z - u) * inv) + b.z;
    o.w = w.w * ((x.w - u) * inv) + b.w;
    *(float4*)&out[(gtok0 + tok) * HID + ln * 4] = o;
  }
}

extern "C" void kernel_launch(void* const* d_in, const int* in_sizes, int n_in,
                              void* d_out, int out_size, void* d_ws, size_t ws_size,
                              hipStream_t stream) {
  const float* h    = (const float*)d_in[0];
  const float* msg  = (const float*)d_in[1];
  const float* mask = (const float*)d_in[2];
  const float* qW   = (const float*)d_in[3];
  const float* qb   = (const float*)d_in[4];
  const float* kW   = (const float*)d_in[5];
  const float* kb   = (const float*)d_in[6];
  const float* vW   = (const float*)d_in[7];
  const float* vb   = (const float*)d_in[8];
  const float* dW   = (const float*)d_in[9];
  const float* db   = (const float*)d_in[10];
  const float* lnw  = (const float*)d_in[11];
  const float* lnb  = (const float*)d_in[12];
  float* out = (float*)d_out;
  _Float16* wsW = (_Float16*)d_ws;        // 4 * 32KB fragment-packed weights

  // Pre-pack the four 128x128 weight matrices into WMMA B-fragment order.
  pack_weights<<<(4 * WFRAG) / 256, 256, 0, stream>>>(qW, kW, vW, dW, wsW);

  const int tokens = in_sizes[0] / HID;   // bz*n = 16384
  const int blocks = tokens / TPB;        // 4096 workgroups
  const size_t smem = 150528;             // ~147 KB dynamic LDS
  softattn_gfx1250<<<blocks, 256, smem, stream>>>(
      h, msg, mask, qb, kb, vb, db, lnw, lnb, wsW, out);
}
